// EqStftSnsePBC_51436528337750
// MI455X (gfx1250) — matrix operand order repaired and checked by reading, not compiled, not run on hardware
//
#include <hip/hip_runtime.h>
#include <cstddef>
#include <cstdint>

// ---------------------------------------------------------------------------
// EqStftSnsePBC on MI455X (gfx1250), wave32 + WMMA + Tensor Data Mover.
//
// FFT-256 = four-step 16x16 factorization:
//   X[k2+16*k1] = sum_n1 W16^{n1 k1} * W256^{n1 k2} * sum_n2 x[n1+16*n2] W16^{n2 k2}
// Each complex 16x16x16 matmul is ONE v_wmma_f32_16x16x32_f16 per real/imag
// output: A = [Ar | Ai] (16x32 f16), B1 = [F16r ; -F16i], B2 = [F16i ; F16r].
// Stage-1 C-layout output == stage-2 A-layout input (F16 symmetric), so the
// only inter-stage work is twiddle + f32->f16 pack (no cross-lane traffic).
//
// Frame input (512 contiguous f32 per array per frame) is staged to LDS with
// TENSOR_LOAD_TO_LDS (TDM), overlapped with per-wave constant setup, waited
// with s_wait_tensorcnt. This replaces 50%-utilized per-lane dword gathers
// with full-cacheline DMA traffic (the input stage dominates HBM bytes).
// ---------------------------------------------------------------------------

typedef __attribute__((ext_vector_type(16))) _Float16     v16h;
typedef __attribute__((ext_vector_type(8)))  float        v8f;
typedef __attribute__((ext_vector_type(4)))  unsigned int u32x4;
typedef __attribute__((ext_vector_type(8)))  unsigned int u32x8;

#define BATCH 2
#define STEPS 4628
#define NFFT  256
#define HOPSZ 216
#define LTOT  999688      // (STEPS-1)*HOPSZ + NFFT
#define LOUT  999648      // LTOT - 2*PADH
#define NMODE 2
#define MTAP  41
#define PADH  20          // M//2
#define LPAD  (NFFT + 2*PADH)   // 296
#define FRMF  (NFFT * NMODE)    // 512 floats per frame per array

static_assert((BATCH * STEPS) % 4 == 0, "one wave per frame, 4 waves per block");
static_assert((STEPS - 1) * HOPSZ + NFFT == LTOT, "length check");

__device__ __forceinline__ v8f wmma16(v16h a, v16h b, v8f c) {
  // 8 args: (neg_a, A, neg_b, B, c_mod, C, reuse_a, reuse_b)
  return __builtin_amdgcn_wmma_f32_16x16x32_f16(false, a, false, b, (short)0, c,
                                                false, false);
}

// ---------------------------------------------------------------------------
// TDM: load `nelem` contiguous 4-byte elements from global to LDS.
// D# per CDNA5 ISA ch.8: group0 = {count=1 | lds_addr | global_addr | type=2},
// group1 = {data_size=4B, tensor_dim0=nelem, tensor_dim1=1, tile_dim0=nelem,
//           tile_dim1=1, stride0=nelem}; groups 2/3 zero (dims 2..4 unused).
// Descriptor values are wave-uniform; TDM issues once per wave, EXEC ignored,
// completion tracked with TENSORcnt.
// ---------------------------------------------------------------------------
__device__ __forceinline__ void tdm_load_1d(const void* gptr, void* lptr,
                                            unsigned int nelem) {
  uint64_t ga = (uint64_t)(uintptr_t)gptr;
  uint32_t la = (uint32_t)(uintptr_t)lptr;        // low 32 bits = LDS offset
  u32x4 g0 = { 1u,                                 // count=1, user descriptor
               la,                                 // lds_addr  (bits 63:32)
               (uint32_t)ga,                       // global_addr[31:0]
               (uint32_t)(ga >> 32) | (2u << 30) };// addr[56:32] | type=2
  u32x8 g1 = { 0x00020000u,                        // data_size = 2 (4 bytes)
               nelem << 16,                        // tensor_dim0[15:0] @ bit48
               0x00010000u,                        // tensor_dim1 = 1  @ bit80
               nelem << 16,                        // tile_dim0 @ bits 127:112
               1u,                                 // tile_dim1 = 1
               nelem,                              // tensor_dim0_stride
               0u, 0u };
  u32x4 gz = { 0u, 0u, 0u, 0u };
  asm volatile("tensor_load_to_lds %0, %1, %2, %3"
               :: "s"(g0), "s"(g1), "s"(gz), "s"(gz)
               : "memory");
}

// Forward FFT-256 of the complex data packed in a1 (A-layout), result in
// C-layout: lane l, reg r  ->  X[(r + 8*(l>=16)) + 16*(l&15)].
__device__ __forceinline__ void fft256(v16h a1, v16h b1, v16h b2,
                                       const float* twr, const float* twi,
                                       v8f& outr, v8f& outi) {
  v8f z = {0.f, 0.f, 0.f, 0.f, 0.f, 0.f, 0.f, 0.f};
  v8f gr = wmma16(a1, b1, z);   // stage 1 real
  v8f gi = wmma16(a1, b2, z);   // stage 1 imag
  v16h a2;
  #pragma unroll
  for (int r = 0; r < 8; ++r) { // twiddle H = G * W256^{n1*k2}, pack to f16
    float hre = gr[r] * twr[r] - gi[r] * twi[r];
    float him = gr[r] * twi[r] + gi[r] * twr[r];
    a2[r]     = (_Float16)hre;
    a2[r + 8] = (_Float16)him;
  }
  outr = wmma16(a2, b1, z);     // stage 2 real
  outi = wmma16(a2, b2, z);     // stage 2 imag
}

__global__ void __launch_bounds__(128)
eqstft_frames_kernel(const float* __restrict__ gxr, const float* __restrict__ gxi,
                     const float* __restrict__ gtask,
                     const float* __restrict__ ghr, const float* __restrict__ ghi,
                     float2* __restrict__ yf) {
  __shared__ float  sFrm[4][2][FRMF];          // TDM-staged frame data (re,im)
  __shared__ float  sh[2][MTAP];               // filter taps (shared per block)
  __shared__ float  sIp[4][LPAD + 8];          // wrap-padded power spectrum
  __shared__ float2 sPhi[4][NFFT];             // phi redistribution
  __shared__ float2 sY[4][NMODE][NFFT];        // per-wave complex staging

  const int tid  = threadIdx.x;
  const int w    = tid >> 5;        // wave id within block
  const int lane = tid & 31;        // wave32
  const int hi   = lane >> 4;       // 0 / 1
  const int lo   = lane & 15;

  const int frame = blockIdx.x * 4 + w;     // exact cover: frames % 4 == 0
  const int b = frame / STEPS;
  const int s = frame % STEPS;
  const int pos0 = s * HOPSZ;

  // ---- kick off the frame DMA first; overlap with constant setup ----------
  const size_t fbase = ((size_t)b * LTOT + (size_t)pos0) * NMODE;
  tdm_load_1d(gxr + fbase, &sFrm[w][0][0], FRMF);
  tdm_load_1d(gxi + fbase, &sFrm[w][1][0], FRMF);

  if (tid < MTAP) { sh[0][tid] = ghr[tid]; sh[1][tid] = ghi[tid]; }

  // ---- constant B operands (DFT-16 matrix, W16 = e^{-2pi i /16}) ----------
  // B layout: elem e, lane l -> B[K][N], K = (e&7) + 8*hi + 16*(e>=8), N = lo.
  // b1 rows: [F16r ; -F16i]   b2 rows: [F16i ; F16r]
  v16h b1, b2;
  #pragma unroll
  for (int e = 0; e < 16; ++e) {
    float ang = 0.39269908169872414f * (float)((((e & 7) + 8 * hi) * lo) & 15);  // 2pi/16
    float c = cosf(ang), sn = sinf(ang);     // F16r = c, F16i = -sn
    b1[e] = (_Float16)((e >= 8) ? sn : c);
    b2[e] = (_Float16)((e >= 8) ? c : -sn);
  }
  // ---- inter-stage twiddles W256^{n1*k2}, C-layout (n1 = r+8*hi, k2 = lo) -
  float twr[8], twi[8];
  #pragma unroll
  for (int r = 0; r < 8; ++r) {
    float ang = 0.02454369260617026f * (float)((r + 8 * hi) * lo);  // 2pi/256
    twr[r] = cosf(ang);
    twi[r] = -sinf(ang);
  }

  const float P = expf(gtask[b * 4 + 0] * 0.2302585093f) * 0.5f; // 10^(t/10)/Nm

  // ---- wait for the TDM transfers, then fence LDS ordering ---------------
  __builtin_amdgcn_s_wait_tensorcnt(0);
  __syncthreads();

  // ---- pass A: forward FFT both modes, accumulate power spectrum ----------
  v8f Xre[NMODE], Xim[NMODE];
  float iacc[8];
  #pragma unroll
  for (int r = 0; r < 8; ++r) iacc[r] = 0.f;

  #pragma unroll
  for (int m = 0; m < NMODE; ++m) {
    v16h a1;   // A layout: elem j -> x[(lo) + 16*(j + 8*hi)], j+8 = imag
    #pragma unroll
    for (int j = 0; j < 8; ++j) {
      int n = lo + 16 * (j + 8 * hi);
      a1[j]     = (_Float16)sFrm[w][0][n * NMODE + m];
      a1[j + 8] = (_Float16)sFrm[w][1][n * NMODE + m];
    }
    fft256(a1, b1, b2, twr, twi, Xre[m], Xim[m]);
    #pragma unroll
    for (int r = 0; r < 8; ++r)
      iacc[r] += Xre[m][r] * Xre[m][r] + Xim[m][r] * Xim[m][r];
  }

  // ---- wrap-padded I into LDS: Ipad[p] = I[(p-20) mod 256] ---------------
  #pragma unroll
  for (int r = 0; r < 8; ++r) {
    int k = (r + 8 * hi) + 16 * lo;
    float v = iacc[r];
    sIp[w][k + PADH] = v;
    if (k < PADH)         sIp[w][k + PADH + NFFT] = v;   // right pad
    if (k >= NFFT - PADH) sIp[w][k - (NFFT - PADH)] = v; // left pad
  }
  __syncthreads();

  // ---- phi[n] = sum_j Ipad[n+j] * h[j]; computed at n = lane + 32*r ------
  // (contiguous lane addressing -> conflict-free LDS reads)
  float phr[8], phim[8];
  #pragma unroll
  for (int r = 0; r < 8; ++r) { phr[r] = 0.f; phim[r] = 0.f; }
  for (int j = 0; j < MTAP; ++j) {
    float hre = sh[0][j], him = sh[1][j];
    #pragma unroll
    for (int r = 0; r < 8; ++r) {
      float Iv = sIp[w][lane + 32 * r + j];
      phr[r]  += Iv * hre;
      phim[r] += Iv * him;
    }
  }
  #pragma unroll
  for (int r = 0; r < 8; ++r) sPhi[w][lane + 32 * r] = make_float2(phr[r], phim[r]);
  __syncthreads();

  // redistribute phi to the C-layout positions, pre-multiplied by P
  float pr[8], pi[8];
  #pragma unroll
  for (int r = 0; r < 8; ++r) {
    float2 ph = sPhi[w][(r + 8 * hi) + 16 * lo];
    pr[r] = P * ph.x;
    pi[r] = P * ph.y;
  }

  // ---- pass B: Y = X*((1-P*phi_i) + i*P*phi_r), ifft = conj(fft(conj))/256
  #pragma unroll
  for (int m = 0; m < NMODE; ++m) {
    #pragma unroll
    for (int r = 0; r < 8; ++r) {
      int k = (r + 8 * hi) + 16 * lo;
      float fr = 1.0f - pi[r];
      float fi = pr[r];
      float yr = Xre[m][r] * fr - Xim[m][r] * fi;
      float yi = Xim[m][r] * fr + Xre[m][r] * fi;
      // store conj(Y)/16 (keeps f16 inputs far from overflow; rescaled below)
      sY[w][m][k] = make_float2(yr * 0.0625f, -yi * 0.0625f);
    }
    __syncthreads();
    v16h a1;
    #pragma unroll
    for (int j = 0; j < 8; ++j) {
      int n = lo + 16 * (j + 8 * hi);
      float2 cv = sY[w][m][n];
      a1[j]     = (_Float16)cv.x;
      a1[j + 8] = (_Float16)cv.y;
    }
    v8f Zr, Zi;
    fft256(a1, b1, b2, twr, twi, Zr, Zi);
    #pragma unroll
    for (int r = 0; r < 8; ++r) {
      int k = (r + 8 * hi) + 16 * lo;
      // yf = conj(Z')*16/256 = conj(Z')*0.0625
      sY[w][m][k] = make_float2(Zr[r] * 0.0625f, -Zi[r] * 0.0625f);
    }
    __syncthreads();
  }

  // ---- coalesced frame write-out: yf[((b*STEPS+s)*256 + n)*2 + m] --------
  float2* dst = yf + (size_t)(b * STEPS + s) * NFFT * NMODE;
  for (int t = lane; t < NFFT * NMODE; t += 32) {
    dst[t] = sY[w][t & 1][t >> 1];
  }
}

// ---------------------------------------------------------------------------
// Overlap-add gather + weight normalization + crop.
// Each output sample p is covered by <=2 frames (hop 216 > overlap 40);
// coverage count is computed analytically (replaces the wsum scatter).
// ---------------------------------------------------------------------------
__global__ void __launch_bounds__(256)
eqstft_gather_kernel(const float2* __restrict__ yf, float* __restrict__ out) {
  const int total = BATCH * LOUT * NMODE;
  int t = blockIdx.x * blockDim.x + threadIdx.x;
  if (t >= total) return;
  int m    = t & 1;
  int rest = t >> 1;
  int ipos = rest % LOUT;
  int b    = rest / LOUT;
  int p    = ipos + PADH;                       // position in full length L

  int s_hi = p / HOPSZ;           if (s_hi > STEPS - 1) s_hi = STEPS - 1;
  int s_lo = (p >= NFFT) ? (p - (NFFT - 1) + HOPSZ - 1) / HOPSZ : 0;

  float accr = 0.f, acci = 0.f;
  int cnt = 0;
  for (int s = s_lo; s <= s_hi; ++s) {
    int n = p - s * HOPSZ;                       // 0 <= n < 256
    float2 v = yf[((size_t)(b * STEPS + s) * NFFT + n) * NMODE + m];
    accr += v.x; acci += v.y; ++cnt;
  }
  float inv = 1.0f / (float)cnt;
  float2* o = (float2*)out;                      // (B, LOUT, NMODE, 2)
  o[(size_t)t] = make_float2(accr * inv, acci * inv);
}

extern "C" void kernel_launch(void* const* d_in, const int* in_sizes, int n_in,
                              void* d_out, int out_size, void* d_ws, size_t ws_size,
                              hipStream_t stream) {
  (void)in_sizes; (void)n_in; (void)out_size; (void)ws_size;
  const float* xr   = (const float*)d_in[0];   // (B, L, Nm) f32
  const float* xi   = (const float*)d_in[1];   // (B, L, Nm) f32
  const float* task = (const float*)d_in[2];   // (B, 4) f32
  const float* hr   = (const float*)d_in[3];   // (41,) f32
  const float* hi   = (const float*)d_in[4];   // (41,) f32

  // workspace: per-frame IFFT outputs, B*STEPS*256*Nm float2 (~36 MB)
  float2* yf = (float2*)d_ws;

  const int frames = BATCH * STEPS;            // 9256, divisible by 4
  eqstft_frames_kernel<<<frames / 4, 128, 0, stream>>>(xr, xi, task, hr, hi, yf);

  const int total = BATCH * LOUT * NMODE;
  eqstft_gather_kernel<<<(total + 255) / 256, 256, 0, stream>>>(yf, (float*)d_out);
}
